// SparseConvNet_584115552642
// MI455X (gfx1250) — matrix operand order repaired and checked by reading, not compiled, not run on hardware
//
#include <hip/hip_runtime.h>
#include <stdint.h>

typedef __attribute__((ext_vector_type(16))) _Float16 v16h;
typedef __attribute__((ext_vector_type(8)))  _Float16 v8h;
typedef __attribute__((ext_vector_type(8)))  float    v8f;
typedef unsigned int v4u __attribute__((ext_vector_type(4)));
typedef int          v8i __attribute__((ext_vector_type(8)));
typedef int          v4i __attribute__((ext_vector_type(4)));

// ---------------------------------------------------------------------------
// Zero fill (zero page needs deterministic zeros every call)
// ---------------------------------------------------------------------------
__global__ void zero_f16_k(_Float16* __restrict__ p, int n)
{
    int i = blockIdx.x * blockDim.x + threadIdx.x;
    if (i < n) p[i] = (_Float16)0.0f;
}

// ---------------------------------------------------------------------------
// BN fold: scale = g * rsqrt(v + eps); bias = b - m*scale
// ---------------------------------------------------------------------------
__global__ void bn_prep_k(const float* __restrict__ b, const float* __restrict__ g,
                          const float* __restrict__ m, const float* __restrict__ v,
                          float* __restrict__ sc, float* __restrict__ bi, int C)
{
    int i = blockIdx.x * blockDim.x + threadIdx.x;
    if (i < C) {
        float s = g[i] * rsqrtf(v[i] + 1e-3f);
        sc[i] = s;
        bi[i] = b[i] - m[i] * s;
    }
}

// ---------------------------------------------------------------------------
// Weight packing: [Co][Cin][3][3][3] fp32 -> fragment-ordered f16 (B-matrix
// 32x16 16-bit layout: lane 0-15 = K0..7,16..23 col N=lane; lane 16-31 = K8..15,24..31).
//  normal: fragment f = (tap*nch + chunk)*(CO/16) + coTile
//          K=chunk*32+(j>=8?16:0)+(lane>=16?8:0)+(j&7)
//  pair (CIN==16): fragment f = p*(CO/16) + coTile covers taps {2p, 2p+1}
//          j<8: tap=2p; j>=8: tap=2p+1; k=(lane>=16?8:0)+(j&7); tap 27 -> zeros
// ---------------------------------------------------------------------------
__global__ void pack_w_k(const float* __restrict__ w, _Float16* __restrict__ out,
                         int CIN, int CO, int nch, int pair, int total)
{
    int idx = blockIdx.x * blockDim.x + threadIdx.x;
    if (idx >= total) return;
    int j    = idx & 15;
    int lane = (idx >> 4) & 31;
    int frag = idx >> 9;
    int nCoT = CO >> 4;
    int coT  = frag % nCoT;
    int tc   = frag / nCoT;
    int co   = coT * 16 + (lane & 15);
    float val;
    if (pair) {
        int tap = 2 * tc + ((j >> 3) & 1);
        int kk  = ((lane & 16) ? 8 : 0) + (j & 7);
        val = (tap < 27) ? w[((long)co * CIN + kk) * 27 + tap] : 0.0f;
    } else {
        int c   = tc % nch;
        int tap = tc / nch;
        int k   = c * 32 + ((j & 8) ? 16 : 0) + ((lane & 16) ? 8 : 0) + (j & 7);
        val = (k < CIN) ? w[((long)co * CIN + k) * 27 + tap] : 0.0f;
    }
    out[idx] = (_Float16)val;
}

// ---------------------------------------------------------------------------
// NCDHW fp32 -> [vox][C] f16 (activation layout for A-fragment loads)
// ---------------------------------------------------------------------------
__global__ void to_vc_f16_k(const float* __restrict__ x, _Float16* __restrict__ o,
                            int V, int C)
{
    long i = (long)blockIdx.x * blockDim.x + threadIdx.x;
    if (i >= (long)V * C) return;
    int c  = (int)(i % C);
    long v = i / C;
    o[i] = (_Float16)x[(long)c * V + v];
}

// ---------------------------------------------------------------------------
// Mask downsample: OR over 3x3x3 stride-2 neighborhood (== conv(m,ones,2)>0)
// ---------------------------------------------------------------------------
__global__ void mask_down_k(const float* __restrict__ mi, float* __restrict__ mo,
                            int Ri, int Ro)
{
    int v = blockIdx.x * blockDim.x + threadIdx.x;
    int V = Ro * Ro * Ro;
    if (v >= V) return;
    int x = v % Ro, t = v / Ro, y = t % Ro, z = t / Ro;
    float r = 0.0f;
    for (int dz = -1; dz <= 1; ++dz)
        for (int dy = -1; dy <= 1; ++dy)
            for (int dx = -1; dx <= 1; ++dx) {
                int iz = 2 * z + dz, iy = 2 * y + dy, ix = 2 * x + dx;
                if (iz >= 0 && iz < Ri && iy >= 0 && iy < Ri && ix >= 0 && ix < Ri)
                    if (mi[((long)iz * Ri + iy) * Ri + ix] > 0.0f) r = 1.0f;
            }
    mo[v] = r;
}

// ---------------------------------------------------------------------------
// Implicit-GEMM 3^3 conv + BN + ReLU + mask.
//  Block = 4 waves; each wave owns 4 M-subtiles (64 voxels) x 16 cout.
//  Per subtile, all tap validity lives in one 27-bit mask and one prefolded
//  base pointer; steady-state A loads use immediate offsets only. B fragments
//  staged per-workgroup into LDS by the TDM. CIN==16 uses tap-pairing.
// ---------------------------------------------------------------------------
template <int CIN, int CO, int STRIDE, int RI, int RO>
__global__ __launch_bounds__(128)
void conv3d_wmma(const _Float16* __restrict__ in, const _Float16* __restrict__ wpk,
                 const float* __restrict__ bnsc, const float* __restrict__ bnbi,
                 const float* __restrict__ mask, const _Float16* __restrict__ zp,
                 _Float16* __restrict__ outh, float* __restrict__ outf)
{
    constexpr int  NCH   = (CIN + 31) / 32;
    constexpr bool PAIR  = (CIN == 16);
    constexpr int  NFRAG = PAIR ? 14 : 27 * NCH;
    constexpr int  NCOT  = CO / 16;
    constexpr int  V     = RO * RO * RO;
    constexpr int  MT    = 4;                   // M-subtiles per wave
    extern __shared__ _Float16 smem[];          // NFRAG * 512 halves

    union A16 { v16h v; v8h h[2]; };

    const int lane = threadIdx.x & 31;
    const int wave = threadIdx.x >> 5;
    const int coT  = blockIdx.y;

    // ---- TDM: stage all B fragments for this coT into LDS (one 2-D tile:
    //      row = 512 halves, row stride = NCOT*512 halves, NFRAG rows) ------
    if (wave == 0) {
        unsigned lds = (unsigned)(uintptr_t)smem;
        unsigned long long ga =
            (unsigned long long)(uintptr_t)(wpk + (size_t)coT * 512);
        // D# group0: count=1 | lds_addr | global_addr | type=2
        v4u g0 = { 1u, lds, (unsigned)ga,
                   ((unsigned)(ga >> 32) & 0x01FFFFFFu) | 0x80000000u };
        // D# group1: data_size=2B; tensor_dim0=512; tensor_dim1=NFRAG;
        //            tile_dim0=512; tile_dim1=NFRAG; stride0=NCOT*512 elems
        v8i g1 = { 0x10000,
                   (int)(512u << 16),
                   (int)((unsigned)NFRAG << 16),
                   (int)(512u << 16),
                   NFRAG,
                   NCOT * 512,
                   0, 0 };
        v4i gz = { 0, 0, 0, 0 };
#if defined(__clang_major__) && (__clang_major__ >= 23)
        v8i gz8 = { 0, 0, 0, 0, 0, 0, 0, 0 };
        __builtin_amdgcn_tensor_load_to_lds(g0, g1, gz, gz, gz8, 0);
#else
        __builtin_amdgcn_tensor_load_to_lds(g0, g1, gz, gz, 0);
#endif
        __builtin_amdgcn_s_wait_tensorcnt(0);
    }
    __syncthreads();

    const int hiHalf = (lane & 16) ? 8 : 0;     // lane-dependent K offset

    // ---- per-subtile state: prefolded base pointer + 27-bit tap validity ---
    const int vbase = (blockIdx.x * 4 + wave) * (16 * MT);
    const _Float16* bp[MT];
    unsigned okm[MT];
#pragma unroll
    for (int t = 0; t < MT; ++t) {
        const int v = vbase + t * 16 + (lane & 15);
        const bool vvb = v < V;
        const int x = v % RO;
        const int q = v / RO;
        const int y = q % RO;
        const int z = q / RO;
        const int sxx = x * STRIDE, syy = y * STRIDE, szz = z * STRIDE;
        bp[t] = in + ((szz * RI + syy) * RI + sxx) * CIN + hiHalf;
        unsigned m = 0;
#pragma unroll
        for (int tap = 0; tap < 27; ++tap) {
            const int dz = tap / 9 - 1, dy = (tap / 3) % 3 - 1, dx = tap % 3 - 1;
            const bool ok = vvb &
                ((unsigned)(szz + dz) < (unsigned)RI) &
                ((unsigned)(syy + dy) < (unsigned)RI) &
                ((unsigned)(sxx + dx) < (unsigned)RI);
            m |= ((unsigned)ok) << tap;
        }
        okm[t] = m;
    }
    const _Float16* zph = zp + hiHalf;

    // branchless tap pointer: 1-bit test + const-offset add + cndmask pair
    auto tapQ = [&](int t, int tap) -> const _Float16* {
        const int dz = tap / 9 - 1, dy = (tap / 3) % 3 - 1, dx = tap % 3 - 1;
        const int off = ((dz * RI + dy) * RI + dx) * CIN;
        return ((okm[t] >> tap) & 1u) ? (bp[t] + off) : zph;
    };

    v8f acc[MT] = {};

    if constexpr (PAIR) {
        // CIN==16: one K=32 fragment covers taps {2p, 2p+1}
#pragma unroll
        for (int p = 0; p < 14; ++p) {
            const int t0 = 2 * p;
            const int t1 = (2 * p + 1 < 27) ? 2 * p + 1 : 2 * p;
            v16h b = *(const v16h*)(smem + p * 512 + lane * 16);
#pragma unroll
            for (int t = 0; t < MT; ++t) {
                A16 a;
                a.h[0] = *(const v8h*)tapQ(t, t0);
                a.h[1] = *(const v8h*)tapQ(t, t1);
                acc[t] = __builtin_amdgcn_wmma_f32_16x16x32_f16(
                    false, a.v, false, b, (short)0, acc[t], false, false);
            }
        }
    } else {
        for (int tap = 0; tap < 27; ++tap) {
            const _Float16* q[MT];
#pragma unroll
            for (int t = 0; t < MT; ++t) q[t] = tapQ(t, tap);
#pragma unroll
            for (int c = 0; c < NCH; ++c) {
                v16h b = *(const v16h*)(smem + (tap * NCH + c) * 512 + lane * 16);
#pragma unroll
                for (int t = 0; t < MT; ++t) {
                    A16 a;
                    a.h[0] = *(const v8h*)(q[t] + c * 32);        // imm offset
                    a.h[1] = *(const v8h*)(q[t] + c * 32 + 16);   // imm offset
                    acc[t] = __builtin_amdgcn_wmma_f32_16x16x32_f16(
                        false, a.v, false, b, (short)0, acc[t], false, false);
                }
            }
        }
    }

    // ---- epilogue: lane holds col co, rows M = r + (lane>=16 ? 8:0) --------
    const int co = coT * 16 + (lane & 15);
    const float sc = bnsc[co];
    const float bi = bnbi[co];
    const int mrow0 = (lane & 16) ? 8 : 0;
    if (vbase + 16 * MT <= V) {                  // uniform fast path
#pragma unroll
        for (int t = 0; t < MT; ++t) {
#pragma unroll
            for (int r = 0; r < 8; ++r) {
                const int v = vbase + t * 16 + mrow0 + r;
                float val = fmaxf(acc[t][r] * sc + bi, 0.0f) * mask[v];
                outh[v * CO + co] = (_Float16)val;
                if (outf) outf[co * V + v] = val;        // NCDHW fp32 export
            }
        }
    } else {                                     // partial tail tile
#pragma unroll
        for (int t = 0; t < MT; ++t) {
#pragma unroll
            for (int r = 0; r < 8; ++r) {
                const int v = vbase + t * 16 + mrow0 + r;
                if (v < V) {
                    float val = fmaxf(acc[t][r] * sc + bi, 0.0f) * mask[v];
                    outh[v * CO + co] = (_Float16)val;
                    if (outf) outf[co * V + v] = val;
                }
            }
        }
    }
}

// ---------------------------------------------------------------------------
// Host-side conv dispatcher (4 waves/block, 256 voxels/block, dynamic LDS)
// ---------------------------------------------------------------------------
static inline void launch_conv(int cin, int co, int st, int ri, int ro,
                               const _Float16* in, const _Float16* wpk,
                               const float* sc, const float* bi, const float* mk,
                               const _Float16* zp, _Float16* oh, float* of,
                               hipStream_t s)
{
    int V = ro * ro * ro;
    dim3 g((V + 255) / 256, co / 16);
    int nfrag = (cin == 16) ? 14 : 27 * (cin / 32);
    size_t sh = (size_t)nfrag * 512 * 2;
#define LC(CI, CC, SS, RRI, RRO)                                               \
    if (cin == CI && co == CC && st == SS && ri == RRI) {                      \
        (void)hipFuncSetAttribute(                                             \
            reinterpret_cast<const void*>(&conv3d_wmma<CI, CC, SS, RRI, RRO>), \
            hipFuncAttributeMaxDynamicSharedMemorySize, (int)sh);              \
        conv3d_wmma<CI, CC, SS, RRI, RRO><<<g, 128, sh, s>>>(in, wpk, sc, bi,  \
                                                             mk, zp, oh, of);  \
        return;                                                                \
    }
    LC(16, 16, 1, 96, 96)   LC(16, 32, 2, 96, 48)  LC(32, 32, 1, 48, 48)
    LC(32, 64, 2, 48, 24)   LC(64, 64, 1, 24, 24)  LC(64, 128, 2, 24, 12)
    LC(128, 128, 1, 12, 12) LC(128, 128, 2, 12, 6) LC(128, 128, 1, 6, 6)
#undef LC
}

extern "C" void kernel_launch(void* const* d_in, const int* in_sizes, int n_in,
                              void* d_out, int out_size, void* d_ws, size_t ws_size,
                              hipStream_t stream)
{
    (void)in_sizes; (void)n_in; (void)out_size; (void)ws_size;

    const int V96 = 96 * 96 * 96, V48 = 48 * 48 * 48;
    const int V24 = 24 * 24 * 24, V12 = 12 * 12 * 12, V6 = 6 * 6 * 6;

    // -------- workspace layout (all chunks but zp 256B-aligned by construction) ----
    char* ws = (char*)d_ws;
    size_t off = 0;
    _Float16* featA = (_Float16*)(ws + off); off += (size_t)16 * V96 * 2; // 28.3MB
    _Float16* featB = (_Float16*)(ws + off); off += (size_t)16 * V96 * 2; // 28.3MB
    _Float16* wpk   = (_Float16*)(ws + off); off += 3815424ull * 2;       // 7.6MB
    float*    bnp   = (float*)(ws + off);    off += 1408ull * 2 * 4;      // 11KB
    float*    m48   = (float*)(ws + off);    off += (size_t)V48 * 4;
    float*    m24   = (float*)(ws + off);    off += (size_t)V24 * 4;
    float*    m12   = (float*)(ws + off);    off += (size_t)V12 * 4;
    float*    m6    = (float*)(ws + off);    off += (size_t)V6 * 4;
    _Float16* zp    = (_Float16*)(ws + off); off += 512ull * 2;           // zero page

    const float* x   = (const float*)d_in[0];
    const float* m96 = (const float*)d_in[1];
    float* outp = (float*)d_out;

    // -------- zero page (must hold zeros every call; ws is not re-poisoned) -
    zero_f16_k<<<2, 256, 0, stream>>>(zp, 512);

    // -------- mask pyramid (depends only on input mask) --------------------
    mask_down_k<<<(V48 + 255) / 256, 256, 0, stream>>>(m96, m48, 96, 48);
    mask_down_k<<<(V24 + 255) / 256, 256, 0, stream>>>(m48, m24, 48, 24);
    mask_down_k<<<(V12 + 255) / 256, 256, 0, stream>>>(m24, m12, 24, 12);
    mask_down_k<<<(V6 + 255) / 256, 256, 0, stream>>>(m12, m6, 12, 6);

    // -------- input fp32 NCDHW -> f16 [vox][16] -----------------------------
    {
        long tot = (long)16 * V96;
        to_vc_f16_k<<<(unsigned)((tot + 255) / 256), 256, 0, stream>>>(x, featA,
                                                                       V96, 16);
    }

    // -------- run the 9 blocks / 17 convs -----------------------------------
    struct Blk { char kind; int ci, co; };
    const Blk blocks[9] = {{'d', 16, 16},  {'s', 16, 32},  {'d', 32, 32},
                           {'s', 32, 64},  {'t', 64, 64},  {'s', 64, 128},
                           {'t', 128, 128},{'s', 128, 128},{'t', 128, 128}};

    int idx = 2;        // next d_in slot (after x, mask)
    int res = 96;       // current spatial resolution
    _Float16* cur = featA;
    _Float16* nxt = featB;
    _Float16* wp  = wpk;
    float*    bp  = bnp;
    long      ooff = 0; // running offset in d_out

    for (int bidx = 0; bidx < 9; ++bidx) {
        const Blk blk = blocks[bidx];
        const int n = (blk.kind == 'd') ? 2 : ((blk.kind == 't') ? 3 : 1);
        const int wbase = idx + 4 * n;
        const int stride = (blk.kind == 's') ? 2 : 1;
        const int ro = (blk.kind == 's') ? res / 2 : res;
        const float* mk = (ro == 96) ? m96 : (ro == 48) ? m48
                        : (ro == 24) ? m24 : (ro == 12) ? m12 : m6;

        for (int i = 0; i < n; ++i) {
            const int cin = (i == 0) ? blk.ci : blk.co;
            const int co  = blk.co;
            // bn dict flattens key-sorted: b, g, m, v
            const float* bnb = (const float*)d_in[idx + 4 * i + 0];
            const float* bng = (const float*)d_in[idx + 4 * i + 1];
            const float* bnm = (const float*)d_in[idx + 4 * i + 2];
            const float* bnv = (const float*)d_in[idx + 4 * i + 3];
            const float* w   = (const float*)d_in[wbase + i];

            float* sc = bp;
            float* bi = bp + co;
            bp += 2 * co;
            bn_prep_k<<<1, co, 0, stream>>>(bnb, bng, bnm, bnv, sc, bi, co);

            const int pair  = (cin == 16) ? 1 : 0;
            const int nch   = (cin >= 32) ? cin / 32 : 1;
            const int nfrag = pair ? 14 : 27 * nch;
            const int total = nfrag * (co / 16) * 512;
            pack_w_k<<<(total + 255) / 256, 256, 0, stream>>>(w, wp, cin, co,
                                                              nch, pair, total);

            const bool isOut = (blk.kind != 's') && (bidx >= 2) && (i == n - 1);
            float* f32o = isOut ? (outp + ooff) : (float*)nullptr;

            launch_conv(cin, co, stride, res, ro, cur, wp, sc, bi, mk, zp,
                        nxt, f32o, stream);

            if (isOut) ooff += (long)co * ro * ro * ro;
            wp += total;
            _Float16* t = cur; cur = nxt; nxt = t;
        }
        idx += 5 * n;
        res = ro;
    }
}
